// FourierKernel2d_39256001085787
// MI455X (gfx1250) — matrix operand order repaired and checked by reading, not compiled, not run on hardware
//
#include <hip/hip_runtime.h>

// FNO2d on MI455X (gfx1250): truncated-DFT GEMM formulation, all heavy math on
// v_wmma_f32_16x16x32_bf16 (wave32). Operand layouts are fragment-native so each
// WMMA operand is two 128-bit loads per lane (global or LDS).

typedef __attribute__((ext_vector_type(16))) __bf16 v16bf;
typedef __attribute__((ext_vector_type(8)))  __bf16 v8bf;
typedef __attribute__((ext_vector_type(8)))  float  v8f;

static __device__ __forceinline__ v8f wmma_bf16(v16bf a, v16bf b, v8f c) {
  return __builtin_amdgcn_wmma_f32_16x16x32_bf16(false, a, false, b, (short)0, c,
                                                 false, false);
}

static __device__ __forceinline__ v8f zero8() {
  v8f z;
#pragma unroll
  for (int i = 0; i < 8; ++i) z[i] = 0.0f;
  return z;
}

// ---- CDNA5 wave32 WMMA fragment layouts (ISA 7.12.2) ----
// A (16M x 32K): lane = hi*16+m; element e -> K = (e&8?16:0) + (hi?8:0) + (e&7).
// Row-major [m][k] tile => per lane two contiguous 8xbf16 runs (2x b128 loads).
static __device__ __forceinline__ v16bf frag_a_vec(const __bf16* base, int stride) {
  int l = threadIdx.x & 31;
  int m = l & 15;
  int kh = (l >> 4) << 3;
  const __bf16* p = base + (size_t)m * stride + kh;
  v8bf lo = *(const v8bf*)p;         // K = kh .. kh+7
  v8bf hi = *(const v8bf*)(p + 16);  // K = 16+kh .. 16+kh+7
  v16bf a;
#pragma unroll
  for (int e = 0; e < 8; ++e) { a[e] = lo[e]; a[8 + e] = hi[e]; }
  return a;
}
// B (32K x 16N): lane = hi*16+n; element e -> K = (e&8?16:0) + (hi?8:0) + (e&7).
// Transposed [n][k] tile => per lane two contiguous 8xbf16 runs (2x b128 loads).
static __device__ __forceinline__ v16bf frag_b_vec(const __bf16* baseT, int strideT) {
  int l = threadIdx.x & 31;
  int n = l & 15;
  int kh = (l >> 4) << 3;
  const __bf16* p = baseT + (size_t)n * strideT + kh;
  v8bf lo = *(const v8bf*)p;
  v8bf hi = *(const v8bf*)(p + 16);
  v16bf b;
#pragma unroll
  for (int e = 0; e < 8; ++e) { b[e] = lo[e]; b[8 + e] = hi[e]; }
  return b;
}
// D (16x16 f32): element r -> M = (hi?8:0)+r, N = lane&15
template <class Fn>
static __device__ __forceinline__ void store_d_fn(v8f d, Fn f) {
  int l = threadIdx.x & 31;
  int n = l & 15;
  int mh = (l >> 4) << 3;
#pragma unroll
  for (int r = 0; r < 8; ++r) f(mh + r, n, d[r]);
}
// Transposed D store dst[n][m]: per-lane 8 contiguous bf16 -> one b128 store.
static __device__ __forceinline__ void store_d_trans(v8f d, __bf16* base, int strideN) {
  int l = threadIdx.x & 31;
  int n = l & 15;
  int mh = (l >> 4) << 3;
  v8bf o;
#pragma unroll
  for (int r = 0; r < 8; ++r) o[r] = (__bf16)d[r];
  *(v8bf*)(base + (size_t)n * strideN + mh) = o;
}

#define S 256
#define NB 16
#define CW 64
#define LDS_STRIDE 72  // padded [y][c] stride to spread LDS banks

// ------------------------------------------------------------------
// Basis tables (bf16), stored fragment-native.
// Byt[j][y] (32x256): y-forward DFT, 1/256.  T[64][512]: x-forward (A, row-major).
// U[512][64]: x-inverse (A, row-major).      Rt[y][k] (256x32): y-inverse irfft (B^T).
__global__ void k_basis(__bf16* Byt, __bf16* T, __bf16* U, __bf16* Rt) {
  const float W0 = 6.28318530717958647692f / 256.0f;
  int t = threadIdx.x;  // 256 threads
  {
    int y = t;
    for (int j = 0; j < 32; ++j) {
      int ky = j & 15;
      float th = (float)((ky * y) & 255) * W0;
      float v = (j < 16) ? __cosf(th) * (1.0f / 256.0f) : -__sinf(th) * (1.0f / 256.0f);
      Byt[j * 256 + y] = (__bf16)v;
    }
  }
  {
    int r = t & 63, q = t >> 6;
    int rr = r & 31;
    int kx = (rr < 16) ? rr : rr + 224;
    for (int xx = q * 64; xx < q * 64 + 64; ++xx) {
      float th = (float)((kx * xx) & 255) * W0;
      float c = __cosf(th) * (1.0f / 256.0f), s = __sinf(th) * (1.0f / 256.0f);
      if (r < 32) { T[r * 512 + xx] = (__bf16)c;    T[r * 512 + 256 + xx] = (__bf16)s; }
      else        { T[r * 512 + xx] = (__bf16)(-s); T[r * 512 + 256 + xx] = (__bf16)c; }
    }
  }
  {
    int x = t;
    for (int r2 = 0; r2 < 32; ++r2) {
      int kx = (r2 < 16) ? r2 : r2 + 224;
      float th = (float)((kx * x) & 255) * W0;
      float c = __cosf(th), s = __sinf(th);
      U[x * 64 + r2] = (__bf16)c;          U[x * 64 + 32 + r2] = (__bf16)(-s);
      U[(256 + x) * 64 + r2] = (__bf16)s;  U[(256 + x) * 64 + 32 + r2] = (__bf16)c;
    }
  }
  {
    int y = t;
    for (int k = 0; k < 32; ++k) {
      int ky = k & 15;
      float th = (float)((ky * y) & 255) * W0;
      float v = (k < 16) ? (ky == 0 ? 1.0f : 2.0f * __cosf(th))
                         : (ky == 0 ? 0.0f : -2.0f * __sinf(th));
      Rt[y * 32 + k] = (__bf16)v;
    }
  }
}

// Prep: complex channel-mix matrix as bf16, per mode, transposed [n][k]:
// B = [[Wr, Wi], [-Wi, Wr]]
__global__ void k_prepw(const float* __restrict__ w1, const float* __restrict__ w2,
                        __bf16* __restrict__ Wt) {
  int mode = blockIdx.x;  // kx*16 + ky
  int kx = mode >> 4, ky = mode & 15;
  const float* W = (kx < 16) ? w1 : w2;
  int mx = kx & 15;
  int t = threadIdx.x;     // 256 threads
  int n = t >> 1;          // 0..127
  int kbase = (t & 1) * 64;
  int o = n & 63, oi = n >> 6;
  __bf16* dst = Wt + ((size_t)mode * 128 + n) * 128 + kbase;
  for (int kk = 0; kk < 64; ++kk) {
    int k = kbase + kk;
    int i = k & 63, ii = k >> 6;
    size_t idx = ((((size_t)i * 64 + o) * 16 + mx) * 16 + ky) * 2;
    float v = (!ii) ? (oi ? W[idx + 1] : W[idx]) : (oi ? W[idx] : -W[idx + 1]);
    dst[kk] = (__bf16)v;
  }
}

__global__ void k_prepc(const float* __restrict__ cw, __bf16* __restrict__ cwt) {
  for (int i = threadIdx.x; i < CW * CW; i += 256) cwt[i] = (__bf16)cw[i];
}

__global__ void k_prepfc(const float* __restrict__ fc1w, __bf16* __restrict__ fc1t) {
  int n = threadIdx.x;  // 256
  for (int k = 0; k < 64; ++k) fc1t[n * 64 + k] = (__bf16)fc1w[k * 256 + n];
}

// ------------------------------------------------------------------
// Lift: h[b][c][x][y] = sum_k x[b,x,y,k]*fc0_w[k,c] + fc0_b[c]
__global__ void k_lift(const float* __restrict__ x, const float* __restrict__ w,
                       const float* __restrict__ bias, __bf16* __restrict__ h) {
  int bx = blockIdx.x;
  int b = bx >> 8, xi = bx & 255, y = threadIdx.x;
  const float* px = x + (((size_t)(b * 256 + xi) * 256) + y) * 3;
  float v0 = px[0], v1 = px[1], v2 = px[2];
#pragma unroll 4
  for (int c = 0; c < CW; ++c) {
    float r = v0 * w[c] + v1 * w[64 + c] + v2 * w[128 + c] + bias[c];
    h[((size_t)(b * CW + c) << 16) + (xi << 8) + y] = (__bf16)r;
  }
}

// ------------------------------------------------------------------
// Stage a 64(c) x 64(y) tile of h into LDS transposed to [y][c] (stride 72).
// Global reads are b128; transpose happens on the LDS store side.
static __device__ __forceinline__ void stage_hT(const __bf16* hb, __bf16* ldsT) {
  int t = threadIdx.x;          // 128 threads
  int c = t & 63, yq = t >> 6;  // y half (0/1)
  const __bf16* src = hb + ((size_t)c << 16) + yq * 32;
#pragma unroll
  for (int i = 0; i < 4; ++i) {
    v8bf v = *(const v8bf*)(src + i * 8);
#pragma unroll
    for (int e = 0; e < 8; ++e) ldsT[(yq * 32 + i * 8 + e) * LDS_STRIDE + c] = v[e];
  }
}

// ------------------------------------------------------------------
// Stage 1: y-DFT. rows=(b,c,x), K=256(y), N=32. Output transposed Gt[bc][j][x]
// so the D store is one b128 per lane and stage 2's B is vector-loadable.
__global__ void k_dft_y(const __bf16* __restrict__ h, const __bf16* __restrict__ Byt,
                        __bf16* __restrict__ Gt) {
  int wv = threadIdx.x >> 5;
  size_t row0 = (size_t)blockIdx.x * 64 + wv * 16;
  int bc = (int)(row0 >> 8);
  int x0 = (int)(row0 & 255);
  const __bf16* A = h + row0 * 256;
  v8f acc0 = zero8(), acc1 = zero8();
  for (int k0 = 0; k0 < 256; k0 += 32) {
    v16bf a = frag_a_vec(A + k0, 256);
    acc0 = wmma_bf16(a, frag_b_vec(Byt + k0, 256), acc0);
    acc1 = wmma_bf16(a, frag_b_vec(Byt + 16 * 256 + k0, 256), acc1);
  }
  __bf16* gb = Gt + (size_t)bc * 8192 + x0;
  store_d_trans(acc0, gb, 256);
  store_d_trans(acc1, gb + 16 * 256, 256);
}

// ------------------------------------------------------------------
// Stage 2: x-DFT. per (b,c): F(64x16) = T(64x512) * [Gr;Gi](512x16).
// Output in mode-major F2[mode][b][k=im*64+c] for stage 3's A.
__global__ void k_dft_x(const __bf16* __restrict__ Gt, const __bf16* __restrict__ T,
                        __bf16* __restrict__ F2) {
  int wv = threadIdx.x >> 5;
  int bc = blockIdx.x;
  int b = bc >> 6, c = bc & 63;
  const __bf16* Gb = Gt + (size_t)bc * 8192;
  int m0 = wv * 16;
  v8f acc = zero8();
  for (int k0 = 0; k0 < 512; k0 += 32) {
    v16bf a = frag_a_vec(T + (size_t)m0 * 512 + k0, 512);
    const __bf16* Bb = (k0 < 256) ? (Gb + k0) : (Gb + 16 * 256 + (k0 - 256));
    acc = wmma_bf16(a, frag_b_vec(Bb, 256), acc);
  }
  store_d_fn(acc, [&](int m, int n, float v) {
    int j = m0 + m;
    int im = j >> 5, kx = j & 31;
    F2[((size_t)(kx * 16 + n) * 16 + b) * 128 + im * 64 + c] = (__bf16)v;
  });
}

// ------------------------------------------------------------------
// Stage 3: per-mode complex channel mix: O(16x128) = F2_mode(16x128) * Wt_mode^T
__global__ void k_modemix(const __bf16* __restrict__ F2, const __bf16* __restrict__ Wt,
                          __bf16* __restrict__ Fmt) {
  int mode = blockIdx.x;
  int kx = mode >> 4, ky = mode & 15;
  int wv = threadIdx.x >> 5;
  int n0 = wv * 32;
  const __bf16* Ab = F2 + (size_t)mode * 2048;   // [b(16)][k(128)]
  const __bf16* Bb = Wt + (size_t)mode * 16384;  // [n(128)][k(128)]
  v8f acc0 = zero8(), acc1 = zero8();
  for (int k0 = 0; k0 < 128; k0 += 32) {
    v16bf a = frag_a_vec(Ab + k0, 128);
    acc0 = wmma_bf16(a, frag_b_vec(Bb + (size_t)n0 * 128 + k0, 128), acc0);
    acc1 = wmma_bf16(a, frag_b_vec(Bb + (size_t)(n0 + 16) * 128 + k0, 128), acc1);
  }
  auto st = [&](int ng, v8f d) {
    store_d_fn(d, [&](int m, int n, float v) {  // m = b
      int nn = n0 + ng + n;
      int o = nn & 63, oi = nn >> 6;
      Fmt[((size_t)(m * CW + o) * 16 + ky) * 64 + oi * 32 + kx] = (__bf16)v;
    });
  };
  st(0, acc0);
  st(16, acc1);
}

// ------------------------------------------------------------------
// Stage 4: inverse-x. per (b,o): Hst(512x16) = U(512x64) * Fmt_bo^T(64x16).
__global__ void k_idft_x(const __bf16* __restrict__ Fmt, const __bf16* __restrict__ U,
                         __bf16* __restrict__ H) {
  int bc = blockIdx.x;
  int wv = threadIdx.x >> 5;  // 8 waves
  const __bf16* Bb = Fmt + (size_t)bc * 1024;  // [ky(16)][j(64)]
  for (int t = 0; t < 4; ++t) {
    int m0 = (wv * 4 + t) * 16;
    v8f acc = zero8();
    for (int k0 = 0; k0 < 64; k0 += 32) {
      v16bf a = frag_a_vec(U + (size_t)m0 * 64 + k0, 64);
      acc = wmma_bf16(a, frag_b_vec(Bb + k0, 64), acc);
    }
    store_d_fn(acc, [&](int m, int n, float v) {
      int r = m0 + m;
      int x = r & 255, half = r >> 8;
      H[((size_t)bc * 256 + x) * 32 + half * 16 + n] = (__bf16)v;
    });
  }
}

// ------------------------------------------------------------------
// Stage 5+6 fused: y-inverse GEMM (K=32) + 1x1 conv GEMM (K=64) + bias (+ReLU).
// The 64x64 h tile is staged once per block into LDS (transposed), so conv-B
// fragments become 2x ds_load_b128 instead of redundant scalar global gathers.
__global__ void k_fuse(const __bf16* __restrict__ H, const __bf16* __restrict__ Rt,
                       const __bf16* __restrict__ hcur, const __bf16* __restrict__ cwt,
                       const float* __restrict__ cb, __bf16* __restrict__ hnext,
                       int do_relu) {
  __shared__ __bf16 ldsT[64 * LDS_STRIDE];  // [y_local(64)][c(64)]
  int id = blockIdx.x;  // ((b*256)+x)*4 + yt
  int yt = id & 3, x = (id >> 2) & 255, b = id >> 10;
  int y0 = yt * 64;
  const __bf16* hb = hcur + ((size_t)(b * CW) << 16) + x * 256 + y0;
  stage_hT(hb, ldsT);
  __syncthreads();
  int wv = threadIdx.x >> 5;
  int o0 = wv * 16;
  v16bf a1 = frag_a_vec(H + ((size_t)(b * CW + o0) * 256 + x) * 32, 256 * 32);
  v16bf a2a = frag_a_vec(cwt + o0 * 64, 64);
  v16bf a2b = frag_a_vec(cwt + o0 * 64 + 32, 64);
  for (int nt = 0; nt < 4; ++nt) {
    int yy = y0 + nt * 16;
    v8f acc = zero8();
    acc = wmma_bf16(a1, frag_b_vec(Rt + (size_t)yy * 32, 32), acc);
    const __bf16* bt = ldsT + (nt * 16) * LDS_STRIDE;  // [n=y][k=c]
    acc = wmma_bf16(a2a, frag_b_vec(bt, LDS_STRIDE), acc);
    acc = wmma_bf16(a2b, frag_b_vec(bt + 32, LDS_STRIDE), acc);
    store_d_fn(acc, [&](int m, int n, float v) {
      float r = v + cb[o0 + m];
      if (do_relu) r = fmaxf(r, 0.0f);
      hnext[((size_t)(b * CW + o0 + m) << 16) + x * 256 + yy + n] = (__bf16)r;
    });
  }
}

// ------------------------------------------------------------------
// Final: out[p] = relu(h_p @ fc1 + b1) @ fc2 + b2.  Block = 64 pixels (one
// contiguous y-window), staged into LDS transposed so A is 2x ds_load_b128.
__global__ void k_fc(const __bf16* __restrict__ h, const __bf16* __restrict__ fc1t,
                     const float* __restrict__ b1, const float* __restrict__ w2f,
                     const float* __restrict__ b2, float* __restrict__ out) {
  __shared__ __bf16 ldsT[64 * LDS_STRIDE];  // [pixel_local(64)][c(64)]
  size_t pb = (size_t)blockIdx.x * 64;
  int b = (int)(pb >> 16);
  int rem = (int)(pb & 65535);
  const __bf16* hb = h + ((size_t)(b * CW) << 16) + rem;
  stage_hT(hb, ldsT);
  __syncthreads();
  int wv = threadIdx.x >> 5;
  size_t p0 = pb + (size_t)wv * 16;
  const __bf16* Arow = ldsT + (wv * 16) * LDS_STRIDE;
  v16bf aA = frag_a_vec(Arow, LDS_STRIDE);
  v16bf aB = frag_a_vec(Arow + 32, LDS_STRIDE);
  int l = threadIdx.x & 31;
  int nl = l & 15, mh = (l >> 4) << 3;
  float acc[8];
#pragma unroll
  for (int r = 0; r < 8; ++r) acc[r] = 0.0f;
  for (int nt = 0; nt < 16; ++nt) {
    v8f d = zero8();
    d = wmma_bf16(aA, frag_b_vec(fc1t + (size_t)(nt * 16) * 64, 64), d);
    d = wmma_bf16(aB, frag_b_vec(fc1t + (size_t)(nt * 16) * 64 + 32, 64), d);
    int ng = nt * 16 + nl;
    float bias = b1[ng], w2v = w2f[ng];
#pragma unroll
    for (int r = 0; r < 8; ++r) acc[r] += fmaxf(d[r] + bias, 0.0f) * w2v;
  }
#pragma unroll
  for (int off = 1; off < 16; off <<= 1)
#pragma unroll
    for (int r = 0; r < 8; ++r) acc[r] += __shfl_xor(acc[r], off, 32);
  if (nl == 0) {
    float bb = b2[0];
    float4 o0v = make_float4(acc[0] + bb, acc[1] + bb, acc[2] + bb, acc[3] + bb);
    float4 o1v = make_float4(acc[4] + bb, acc[5] + bb, acc[6] + bb, acc[7] + bb);
    *(float4*)(out + p0 + mh) = o0v;
    *(float4*)(out + p0 + mh + 4) = o1v;
  }
}

// ------------------------------------------------------------------
extern "C" void kernel_launch(void* const* d_in, const int* in_sizes, int n_in,
                              void* d_out, int out_size, void* d_ws, size_t ws_size,
                              hipStream_t stream) {
  (void)in_sizes; (void)n_in; (void)out_size; (void)ws_size;
  const float* x     = (const float*)d_in[0];
  const float* fc0_w = (const float*)d_in[1];
  const float* fc0_b = (const float*)d_in[2];
  const float* w1    = (const float*)d_in[3];
  const float* w2    = (const float*)d_in[4];
  const float* cw    = (const float*)d_in[5];
  const float* cb    = (const float*)d_in[6];
  const float* fc1_w = (const float*)d_in[7];
  const float* fc1_b = (const float*)d_in[8];
  const float* fc2_w = (const float*)d_in[9];
  const float* fc2_b = (const float*)d_in[10];
  float* out = (float*)d_out;

  char* ws = (char*)d_ws;
  size_t off = 0;
  const size_t HSZ = (size_t)NB * CW * S * S * sizeof(__bf16);  // 134 MB
  __bf16* hA  = (__bf16*)(ws + off); off += HSZ;
  __bf16* hB  = (__bf16*)(ws + off); off += HSZ;
  __bf16* Gt  = (__bf16*)(ws + off); off += (size_t)NB * CW * 32 * 256 * 2;   // 16.8 MB
  __bf16* F2  = (__bf16*)(ws + off); off += (size_t)512 * 16 * 128 * 2;       // 2 MB
  __bf16* Fmt = (__bf16*)(ws + off); off += (size_t)NB * CW * 16 * 64 * 2;    // 2 MB
  __bf16* H   = (__bf16*)(ws + off); off += (size_t)NB * CW * 256 * 32 * 2;   // 16.8 MB
  __bf16* Wt  = (__bf16*)(ws + off); off += (size_t)512 * 128 * 128 * 2;      // 16.8 MB
  __bf16* Byt = (__bf16*)(ws + off); off += 32 * 256 * 2;
  __bf16* T   = (__bf16*)(ws + off); off += 64 * 512 * 2;
  __bf16* U   = (__bf16*)(ws + off); off += 512 * 64 * 2;
  __bf16* Rt  = (__bf16*)(ws + off); off += 256 * 32 * 2;
  __bf16* cwt = (__bf16*)(ws + off); off += 64 * 64 * 2;
  __bf16* fc1t= (__bf16*)(ws + off); off += 256 * 64 * 2;

  k_basis<<<1, 256, 0, stream>>>(Byt, T, U, Rt);
  k_prepfc<<<1, 256, 0, stream>>>(fc1_w, fc1t);
  k_lift<<<NB * S, 256, 0, stream>>>(x, fc0_w, fc0_b, hA);

  __bf16* cur = hA;
  __bf16* nxt = hB;
  for (int blk = 0; blk < 4; ++blk) {
    const float* w1b = w1 + (size_t)blk * CW * CW * 16 * 16 * 2;
    const float* w2b = w2 + (size_t)blk * CW * CW * 16 * 16 * 2;
    k_prepw<<<512, 256, 0, stream>>>(w1b, w2b, Wt);
    k_prepc<<<1, 256, 0, stream>>>(cw + (size_t)blk * CW * CW, cwt);
    k_dft_y<<<(NB * CW * S) / 64, 128, 0, stream>>>(cur, Byt, Gt);
    k_dft_x<<<NB * CW, 128, 0, stream>>>(Gt, T, F2);
    k_modemix<<<512, 128, 0, stream>>>(F2, Wt, Fmt);
    k_idft_x<<<NB * CW, 256, 0, stream>>>(Fmt, U, H);
    k_fuse<<<NB * S * 4, 128, 0, stream>>>(H, Rt, cur, cwt, cb + (size_t)blk * CW,
                                           nxt, blk < 3 ? 1 : 0);
    __bf16* t = cur; cur = nxt; nxt = t;
  }
  k_fc<<<(NB * S * S) / 64, 128, 0, stream>>>(cur, fc1t, fc1_b, fc2_w, fc2_b, out);
}
